// RSSM_77721728189134
// MI455X (gfx1250) — matrix operand order repaired and checked by reading, not compiled, not run on hardware
//
#include <hip/hip_runtime.h>
#include <stdint.h>

// ---------------- constants from the reference ----------------
#define HH      16
#define BB      2048
#define ACT     10
#define DETER   512
#define HIDDEN  512
#define STOCH   32
#define CLASSES 32
#define SC      (STOCH * CLASSES)   // 1024
#define XKIN    (SC + ACT)          // 1034
#define XK      1056                // 1034 padded to multiple of 32
#define G3      (3 * DETER)         // 1536
#define OUTW    (DETER + SC)        // 1536
#define LN_EPS  1e-5f

typedef __bf16 bf16_t;
typedef __attribute__((ext_vector_type(16))) __bf16 v16bf;
typedef __attribute__((ext_vector_type(8)))  __bf16 v8bf;
typedef __attribute__((ext_vector_type(8)))  float  v8f;

#define WM 4   // 16-row tiles per wave (M)
#define WN 2   // 16-col tiles per wave (N)
// Block: 8 waves = 4 (M) x 2 (N)  ->  block tile = 256 (M) x 64 (N)

// ---------------- WMMA NT GEMM:  C[M,N] = A[M,K] * W[N,K]^T + bias ----------------
// A, W row-major bf16 (K contiguous), C fp32. Compile-time strides so every load
// in the K loop is a global_load_b128 with an immediate offset from 2 base regs.
// Each wave computes a 64x32 macro-tile: 8 WMMAs per 32-K step on 12 b128 loads.
template <int LDA, int LDB, int LDC, int K>
__global__ __launch_bounds__(256) void gemm_bf16_nt(
    const bf16_t* __restrict__ A, const bf16_t* __restrict__ W,
    const float*  __restrict__ bias, float* __restrict__ C)
{
    const int lane  = threadIdx.x & 31;
    const int wave  = threadIdx.x >> 5;
    const int waveM = wave >> 1;                 // 0..3
    const int waveN = wave & 1;                  // 0..1
    const int tileM0 = blockIdx.y * (WM * 16 * 4) + waveM * (WM * 16);
    const int tileN0 = blockIdx.x * (WN * 16 * 2) + waveN * (WN * 16);
    const int mr    = lane & 15;   // A-frag row / B-frag col / C col
    const int hi    = lane >> 4;

    // A 16x32 fragment: lane holds K = {8*hi..8*hi+7} and {16+8*hi..16+8*hi+7}
    const bf16_t* pa = A + (size_t)(tileM0 + mr) * LDA + hi * 8;
    // B 32x16 fragment: lane n holds K = {8*hi..8*hi+7} and {16+8*hi..16+8*hi+7}
    const bf16_t* pb = W + (size_t)(tileN0 + mr) * LDB + hi * 8;

    v8f acc[WM][WN];
#pragma unroll
    for (int m = 0; m < WM; ++m)
#pragma unroll
        for (int n = 0; n < WN; ++n)
            acc[m][n] = (v8f){};

#pragma unroll 2
    for (int k = 0; k < K; k += 32) {
        v16bf av[WM], bv[WN];
#pragma unroll
        for (int m = 0; m < WM; ++m) {
            v8bf a0 = *(const v8bf*)(pa + m * 16 * LDA + k);
            v8bf a1 = *(const v8bf*)(pa + m * 16 * LDA + k + 16);
            av[m] = __builtin_shufflevector(a0, a1,
                     0,1,2,3,4,5,6,7,8,9,10,11,12,13,14,15);
        }
#pragma unroll
        for (int n = 0; n < WN; ++n) {
            v8bf b0 = *(const v8bf*)(pb + n * 16 * LDB + k);
            v8bf b1 = *(const v8bf*)(pb + n * 16 * LDB + k + 16);
            bv[n] = __builtin_shufflevector(b0, b1,
                     0,1,2,3,4,5,6,7,8,9,10,11,12,13,14,15);
        }
#pragma unroll
        for (int m = 0; m < WM; ++m)
#pragma unroll
            for (int n = 0; n < WN; ++n)
                acc[m][n] = __builtin_amdgcn_wmma_f32_16x16x32_bf16(
                    false, av[m], false, bv[n], (short)0, acc[m][n], false, false);
    }

    const int m0 = hi * 8;   // C: lane holds col n, rows v + 8*hi
#pragma unroll
    for (int m = 0; m < WM; ++m) {
#pragma unroll
        for (int n = 0; n < WN; ++n) {
            const float bsv = bias[tileN0 + n * 16 + mr];
            float* Cp = C + (size_t)(tileM0 + m * 16 + m0) * LDC + tileN0 + n * 16 + mr;
#pragma unroll
            for (int v = 0; v < 8; ++v)
                Cp[v * LDC] = acc[m][n][v] + bsv;
        }
    }
}

// ---------------- pointwise helpers ----------------
__device__ __forceinline__ float sigmoidf_(float x) { return 1.0f / (1.0f + __expf(-x)); }

__device__ __forceinline__ uint32_t mix32(uint32_t x) {
    x ^= x >> 16; x *= 0x7feb352du;
    x ^= x >> 15; x *= 0x846ca68bu;
    x ^= x >> 16; return x;
}

// GRU gate fusion: deter' = (1-z)*n + z*h ; also emit bf16 copy + out[t,:,0:512]
__global__ void gru_gate_kernel(const float* __restrict__ gx, const float* __restrict__ gh,
                                float* __restrict__ deter, bf16_t* __restrict__ deter_bf,
                                float* __restrict__ out_t)
{
    int i = blockIdx.x * blockDim.x + threadIdx.x;      // [0, BB*DETER)
    int b = i >> 9, j = i & 511;
    size_t base = (size_t)b * G3 + j;
    float xr = gx[base], xz = gx[base + 512], xn = gx[base + 1024];
    float hr = gh[base], hz = gh[base + 512], hn = gh[base + 1024];
    float r = sigmoidf_(xr + hr);
    float z = sigmoidf_(xz + hz);
    float n = tanhf(xn + r * hn);
    float h = deter[i];
    float d = (1.0f - z) * n + z * h;
    deter[i]    = d;
    deter_bf[i] = (bf16_t)d;
    out_t[(size_t)b * OUTW + j] = d;
}

// LayerNorm + SiLU over rows of 512; one wave per row, lane owns 16 strided elems.
__global__ __launch_bounds__(256) void ln_silu_kernel(
    const float* __restrict__ h1, const float* __restrict__ g,
    const float* __restrict__ beta, bf16_t* __restrict__ h1b)
{
    int lane = threadIdx.x & 31;
    int row  = blockIdx.x * 8 + (threadIdx.x >> 5);
    const float* p = h1 + (size_t)row * HIDDEN;
    float vals[16], s = 0.f, s2 = 0.f;
#pragma unroll
    for (int i = 0; i < 16; ++i) {
        float v = p[lane + i * 32];
        vals[i] = v; s += v; s2 += v * v;
    }
#pragma unroll
    for (int m = 16; m >= 1; m >>= 1) {
        s  += __shfl_xor(s,  m, 32);
        s2 += __shfl_xor(s2, m, 32);
    }
    float mu  = s * (1.0f / HIDDEN);
    float var = s2 * (1.0f / HIDDEN) - mu * mu;
    float inv = rsqrtf(var + LN_EPS);
    bf16_t* q = h1b + (size_t)row * HIDDEN;
#pragma unroll
    for (int i = 0; i < 16; ++i) {
        int c = lane + i * 32;
        float y  = (vals[i] - mu) * inv * g[c] + beta[c];
        float sl = y * (1.0f / (1.0f + __expf(-y)));
        q[c] = (bf16_t)sl;
    }
}

// Gumbel-max categorical sample -> onehot (straight-through forward == onehot).
// Writes onehot to out[t,:,512:] (fp32) and to next-step x buffer (bf16).
__global__ void sample_kernel(const float* __restrict__ logits,
                              float* __restrict__ out_t, bf16_t* __restrict__ xb, int t)
{
    int i = blockIdx.x * blockDim.x + threadIdx.x;      // [0, BB*STOCH)
    int b = i >> 5, s = i & 31;
    const float* lp = logits + (size_t)b * SC + s * 32;
    uint32_t seed = (uint32_t)t * 0x9E3779B9u ^ (uint32_t)i * 0x85EBCA6Bu;
    int best = 0; float bestv = -3.4e38f;
#pragma unroll
    for (int c = 0; c < 32; ++c) {
        uint32_t r = mix32(seed + (uint32_t)c * 0xC2B2AE35u);
        float u = (float)(r >> 8) * (1.0f / 16777216.0f) + (0.5f / 16777216.0f);
        float gmb = -__logf(-__logf(u));
        float v = lp[c] + gmb;
        if (v > bestv) { bestv = v; best = c; }
    }
    float*  op = out_t + (size_t)b * OUTW + DETER + s * 32;
    bf16_t* xp = xb + (size_t)b * XK + s * 32;
#pragma unroll
    for (int c = 0; c < 32; ++c) {
        float v = (c == best) ? 1.0f : 0.0f;
        op[c] = v;
        xp[c] = (bf16_t)v;
    }
}

// fp32 -> bf16 weight copy with K-padding to kd (zeros in pad region)
__global__ void cvt_pad_kernel(const float* __restrict__ src, bf16_t* __restrict__ dst,
                               int rows, int ks, int kd)
{
    long long i = (long long)blockIdx.x * blockDim.x + threadIdx.x;
    if (i >= (long long)rows * kd) return;
    int r = (int)(i / kd), k = (int)(i % kd);
    dst[i] = (k < ks) ? (bf16_t)src[(size_t)r * ks + k] : (bf16_t)0.0f;
}

// x buffer init: stoch0 in [0,1024), zeros in [1024,1056) (action filled per step)
__global__ void init_x_kernel(const float* __restrict__ stoch0, bf16_t* __restrict__ xb)
{
    long long i = (long long)blockIdx.x * blockDim.x + threadIdx.x;
    if (i >= (long long)BB * XK) return;
    int b = (int)(i / XK), k = (int)(i % XK);
    xb[i] = (k < SC) ? (bf16_t)stoch0[(size_t)b * SC + k] : (bf16_t)0.0f;
}

__global__ void init_deter_kernel(const float* __restrict__ d0,
                                  float* __restrict__ df, bf16_t* __restrict__ db)
{
    int i = blockIdx.x * blockDim.x + threadIdx.x;
    if (i >= BB * DETER) return;
    float v = d0[i];
    df[i] = v; db[i] = (bf16_t)v;
}

__global__ void pack_action_kernel(const float* __restrict__ act_t, bf16_t* __restrict__ xb)
{
    int i = blockIdx.x * blockDim.x + threadIdx.x;
    if (i >= BB * ACT) return;
    int b = i / ACT, k = i % ACT;
    xb[(size_t)b * XK + SC + k] = (bf16_t)act_t[i];
}

// ---------------- host orchestration ----------------
extern "C" void kernel_launch(void* const* d_in, const int* in_sizes, int n_in,
                              void* d_out, int out_size, void* d_ws, size_t ws_size,
                              hipStream_t stream)
{
    const float* action = (const float*)d_in[0];
    const float* deter0 = (const float*)d_in[1];
    const float* stoch0 = (const float*)d_in[2];
    const float* W_ih   = (const float*)d_in[3];
    const float* b_ih   = (const float*)d_in[4];
    const float* W_hh   = (const float*)d_in[5];
    const float* b_hh   = (const float*)d_in[6];
    const float* pW1    = (const float*)d_in[7];
    const float* pb1    = (const float*)d_in[8];
    const float* pg     = (const float*)d_in[9];
    const float* pbeta  = (const float*)d_in[10];
    const float* pW2    = (const float*)d_in[11];
    const float* pb2    = (const float*)d_in[12];
    float* out = (float*)d_out;

    char* ws = (char*)d_ws;
    size_t off = 0;
    auto alloc = [&](size_t bytes) -> char* {
        char* p = ws + off;
        off += (bytes + 255) & ~(size_t)255;
        return p;
    };
    bf16_t* Wih_b = (bf16_t*)alloc((size_t)G3 * XK * 2);
    bf16_t* Whh_b = (bf16_t*)alloc((size_t)G3 * DETER * 2);
    bf16_t* W1_b  = (bf16_t*)alloc((size_t)HIDDEN * DETER * 2);
    bf16_t* W2_b  = (bf16_t*)alloc((size_t)SC * HIDDEN * 2);
    bf16_t* xb    = (bf16_t*)alloc((size_t)BB * XK * 2);
    float*  deter = (float*) alloc((size_t)BB * DETER * 4);
    bf16_t* det_b = (bf16_t*)alloc((size_t)BB * DETER * 2);
    float*  gx    = (float*) alloc((size_t)BB * G3 * 4);
    float*  gh    = (float*) alloc((size_t)BB * G3 * 4);
    float*  h1    = (float*) alloc((size_t)BB * HIDDEN * 4);
    bf16_t* h1b   = (bf16_t*)alloc((size_t)BB * HIDDEN * 2);
    float*  lg    = (float*) alloc((size_t)BB * SC * 4);

    const int T = 256;
    auto nb = [](long long n, int t) { return (unsigned)((n + t - 1) / t); };

    // one-time per launch: bf16 (padded) weights + state init
    cvt_pad_kernel<<<nb((long long)G3 * XK, T), T, 0, stream>>>(W_ih, Wih_b, G3, XKIN, XK);
    cvt_pad_kernel<<<nb((long long)G3 * DETER, T), T, 0, stream>>>(W_hh, Whh_b, G3, DETER, DETER);
    cvt_pad_kernel<<<nb((long long)HIDDEN * DETER, T), T, 0, stream>>>(pW1, W1_b, HIDDEN, DETER, DETER);
    cvt_pad_kernel<<<nb((long long)SC * HIDDEN, T), T, 0, stream>>>(pW2, W2_b, SC, HIDDEN, HIDDEN);
    init_x_kernel<<<nb((long long)BB * XK, T), T, 0, stream>>>(stoch0, xb);
    init_deter_kernel<<<nb((long long)BB * DETER, T), T, 0, stream>>>(deter0, deter, det_b);

    const dim3 blk(256);
    // block tile: 256 (M) x 64 (N); M = 2048 -> grid.y = 8
    for (int t = 0; t < HH; ++t) {
        float* out_t = out + (size_t)t * BB * OUTW;
        pack_action_kernel<<<nb((long long)BB * ACT, T), T, 0, stream>>>(
            action + (size_t)t * BB * ACT, xb);
        // gx = x @ W_ih^T + b_ih        [2048 x 1536], K = 1056
        gemm_bf16_nt<XK, XK, G3, XK>
            <<<dim3(G3 / 64, BB / 256), blk, 0, stream>>>(xb, Wih_b, b_ih, gx);
        // gh = h @ W_hh^T + b_hh        [2048 x 1536], K = 512
        gemm_bf16_nt<DETER, DETER, G3, DETER>
            <<<dim3(G3 / 64, BB / 256), blk, 0, stream>>>(det_b, Whh_b, b_hh, gh);
        gru_gate_kernel<<<nb((long long)BB * DETER, T), T, 0, stream>>>(
            gx, gh, deter, det_b, out_t);
        // h1 = deter @ pW1^T + pb1      [2048 x 512], K = 512
        gemm_bf16_nt<DETER, DETER, HIDDEN, DETER>
            <<<dim3(HIDDEN / 64, BB / 256), blk, 0, stream>>>(det_b, W1_b, pb1, h1);
        ln_silu_kernel<<<BB / 8, blk, 0, stream>>>(h1, pg, pbeta, h1b);
        // logits = h1 @ pW2^T + pb2     [2048 x 1024], K = 512
        gemm_bf16_nt<HIDDEN, HIDDEN, SC, HIDDEN>
            <<<dim3(SC / 64, BB / 256), blk, 0, stream>>>(h1b, W2_b, pb2, lg);
        sample_kernel<<<nb((long long)BB * STOCH, T), T, 0, stream>>>(lg, out_t, xb, t);
    }
}
